// Attention_40810779246711
// MI455X (gfx1250) — matrix operand order repaired and checked by reading, not compiled, not run on hardware
//
#include <hip/hip_runtime.h>

typedef __bf16 bf16;
typedef __attribute__((ext_vector_type(16))) __bf16 v16bf;
typedef __attribute__((ext_vector_type(8)))  __bf16 v8bf;
typedef __attribute__((ext_vector_type(8)))  float  v8f;

#define B_     4
#define C_     384
#define C3_    1152
#define HEADS_ 8
#define HD_    48
#define H_     96
#define W_     96
#define HW_    9216
#define EPS_   1e-12f

// ---------------- WMMA helpers (bf16 in, f32 accumulate) ----------------

static __device__ __forceinline__ v8f wmma_bf16(v16bf a, v16bf b, v8f c) {
  // v_wmma_f32_16x16x32_bf16
  return __builtin_amdgcn_wmma_f32_16x16x32_bf16(false, a, false, b, (short)0, c,
                                                 false, false);
}

// A-matrix 16x32 bf16 fragment, row-major source [*, K].
// lane (m = lane&15, h = lane>>4): elems 0..7 -> K = k0+8h+e ; 8..15 -> K = k0+16+8h+(e-8)
static __device__ __forceinline__ v16bf load_a_frag(const bf16* __restrict__ row,
                                                    int k0, int h) {
  const bf16* p = row + k0 + 8 * h;
  v8bf lo = *(const v8bf*)(p);
  v8bf hi = *(const v8bf*)(p + 16);
  v16bf f;
#pragma unroll
  for (int i = 0; i < 8; ++i) { f[i] = lo[i]; f[i + 8] = hi[i]; }
  return f;
}

// B-matrix 32x16 bf16 fragment from B^T row-major [N, K].
// lane (n = lane&15, h = lane>>4): elems 0..7 -> K = k0+8h+e ; 8..15 -> K = k0+16+8h+(e-8)
static __device__ __forceinline__ v16bf load_b_frag(const bf16* __restrict__ row,
                                                    int k0, int h) {
  const bf16* p = row + k0 + 8 * h;
  v8bf lo = *(const v8bf*)(p);
  v8bf hi = *(const v8bf*)(p + 16);
  v16bf f;
#pragma unroll
  for (int i = 0; i < 8; ++i) { f[i] = lo[i]; f[i + 8] = hi[i]; }
  return f;
}

// ---------------- conversion / transpose kernels ----------------

// in: f32 [B][C][HW] -> out: bf16 [B][HW][C]   (B^T layout for GEMM)
__global__ void cvt_transpose(const float* __restrict__ in, bf16* __restrict__ out,
                              int C, int HW, long total) {
  long idx = (long)blockIdx.x * blockDim.x + threadIdx.x;
  if (idx >= total) return;
  int hw = (int)(idx % HW);
  int c  = (int)((idx / HW) % C);
  int b  = (int)(idx / ((long)HW * C));
  out[((size_t)b * HW + hw) * C + c] = (bf16)in[idx];
}

__global__ void cvt_cast(const float* __restrict__ in, bf16* __restrict__ out, long total) {
  long idx = (long)blockIdx.x * blockDim.x + threadIdx.x;
  if (idx >= total) return;
  out[idx] = (bf16)in[idx];
}

// ---------------- big GEMM: C[b] = A (MxK) * B[b] (KxN), B given transposed ----------------
// 32x32 output tile per wave, 8 waves / 256-thread block.
__global__ void gemm_bf16(const bf16* __restrict__ A, const bf16* __restrict__ Bt,
                          float* __restrict__ C, int M, int N, int K,
                          long strideBt, long strideC) {
  int b = blockIdx.y;
  Bt += (size_t)b * strideBt;
  C  += (size_t)b * strideC;
  int wave = blockIdx.x * (blockDim.x >> 5) + (threadIdx.x >> 5);
  int lane = threadIdx.x & 31;
  int tiles_m = M >> 5, tiles_n = N >> 5;
  if (wave >= tiles_m * tiles_n) return;
  int tm = wave / tiles_n, tn = wave % tiles_n;
  int m0 = tm << 5, n0 = tn << 5;
  int l15 = lane & 15, h = lane >> 4;

  const bf16* a0 = A  + (size_t)(m0 + l15) * K;
  const bf16* a1 = A  + (size_t)(m0 + 16 + l15) * K;
  const bf16* b0 = Bt + (size_t)(n0 + l15) * K;
  const bf16* b1 = Bt + (size_t)(n0 + 16 + l15) * K;

  v8f acc00 = {}, acc01 = {}, acc10 = {}, acc11 = {};
  for (int k0 = 0; k0 < K; k0 += 32) {
    v16bf fa0 = load_a_frag(a0, k0, h);
    v16bf fa1 = load_a_frag(a1, k0, h);
    v16bf fb0 = load_b_frag(b0, k0, h);
    v16bf fb1 = load_b_frag(b1, k0, h);
    acc00 = wmma_bf16(fa0, fb0, acc00);
    acc01 = wmma_bf16(fa0, fb1, acc01);
    acc10 = wmma_bf16(fa1, fb0, acc10);
    acc11 = wmma_bf16(fa1, fb1, acc11);
  }
#pragma unroll
  for (int j = 0; j < 8; ++j) {
    int mA = m0 + j + 8 * h, mB = m0 + 16 + j + 8 * h;
    C[(size_t)mA * N + n0 + l15]      = acc00[j];
    C[(size_t)mA * N + n0 + 16 + l15] = acc01[j];
    C[(size_t)mB * N + n0 + l15]      = acc10[j];
    C[(size_t)mB * N + n0 + 16 + l15] = acc11[j];
  }
}

// ---------------- depthwise 3x3 conv, pad=1 ----------------
__global__ void dwconv3x3(const float* __restrict__ in, const float* __restrict__ w,
                          float* __restrict__ out) {
  long idx = (long)blockIdx.x * blockDim.x + threadIdx.x;
  long total = (long)B_ * C3_ * HW_;
  if (idx >= total) return;
  int x = (int)(idx % W_);
  int y = (int)((idx / W_) % H_);
  int c = (int)((idx / HW_) % C3_);
  const float* base = in + (idx - y * (long)W_ - x);  // start of this (b,c) plane
  const float* wp = w + (size_t)c * 9;
  float s = 0.f;
#pragma unroll
  for (int dy = 0; dy < 3; ++dy) {
    int yy = y + dy - 1;
    if (yy < 0 || yy >= H_) continue;
#pragma unroll
    for (int dx = 0; dx < 3; ++dx) {
      int xx = x + dx - 1;
      if (xx < 0 || xx >= W_) continue;
      s += base[yy * W_ + xx] * wp[dy * 3 + dx];
    }
  }
  out[idx] = s;
}

// ---------------- row L2 norms for q,k (reduce over HW) ----------------
__global__ void rownorm(const float* __restrict__ qkv_dw, float* __restrict__ invn) {
  __shared__ float s[256];
  int r = blockIdx.x;  // 0 .. 2*B*HEADS*HD-1   (t slowest)
  int c    = r % HD_;
  int head = (r / HD_) % HEADS_;
  int b    = (r / (HD_ * HEADS_)) % B_;
  int t    = r / (HD_ * HEADS_ * B_);
  const float* row = qkv_dw + ((size_t)b * C3_ + t * C_ + head * HD_ + c) * HW_;
  float acc = 0.f;
  for (int i = threadIdx.x; i < HW_; i += 256) { float x = row[i]; acc += x * x; }
  s[threadIdx.x] = acc; __syncthreads();
  for (int off = 128; off > 0; off >>= 1) {
    if ((int)threadIdx.x < off) s[threadIdx.x] += s[threadIdx.x + off];
    __syncthreads();
  }
  if (threadIdx.x == 0) invn[r] = 1.0f / fmaxf(sqrtf(s[0]), EPS_);
}

// ---------------- pack normalized q,k as bf16 rows [bh][hd][HW] ----------------
__global__ void pack_qk(const float* __restrict__ qkv_dw, const float* __restrict__ invn,
                        bf16* __restrict__ qbf, bf16* __restrict__ kbf) {
  long idx = (long)blockIdx.x * blockDim.x + threadIdx.x;
  long total = 2L * B_ * HEADS_ * HD_ * HW_;
  if (idx >= total) return;
  int n = (int)(idx % HW_);
  long r = idx / HW_;
  int c    = (int)(r % HD_);
  int head = (int)((r / HD_) % HEADS_);
  int b    = (int)((r / (HD_ * HEADS_)) % B_);
  int t    = (int)(r / (HD_ * HEADS_ * B_));
  float v = qkv_dw[((size_t)b * C3_ + t * C_ + head * HD_ + c) * HW_ + n] * invn[r];
  bf16* out = t ? kbf : qbf;
  out[((size_t)(b * HEADS_ + head) * HD_ + c) * HW_ + n] = (bf16)v;
}

// ---------------- pack v transposed bf16 [bh][HW][64] (K padded 48->64) ----------------
__global__ void pack_vT(const float* __restrict__ qkv_dw, bf16* __restrict__ vT) {
  long idx = (long)blockIdx.x * blockDim.x + threadIdx.x;
  long total = (long)B_ * HEADS_ * HW_ * 64;
  if (idx >= total) return;
  int c = (int)(idx & 63);
  long rest = idx >> 6;
  int n  = (int)(rest % HW_);
  int bh = (int)(rest / HW_);
  int b = bh >> 3, head = bh & 7;
  float v = (c < HD_)
      ? qkv_dw[((size_t)b * C3_ + 2 * C_ + head * HD_ + c) * HW_ + n] : 0.f;
  vT[idx] = (bf16)v;
}

// ---------------- attn = q * k^T per (b,head): M=N=48, K=9216 ----------------
// 9 waves (288 threads): one 16x16 tile each; grid = B*HEADS.
__global__ void attn_gemm(const bf16* __restrict__ qbf, const bf16* __restrict__ kbf,
                          float* __restrict__ attn) {
  int bh = blockIdx.x;
  int wave = threadIdx.x >> 5, lane = threadIdx.x & 31;
  int tm = wave / 3, tn = wave % 3;
  int l15 = lane & 15, h = lane >> 4;
  const bf16* arow = qbf + ((size_t)bh * HD_ + tm * 16 + l15) * HW_;
  const bf16* brow = kbf + ((size_t)bh * HD_ + tn * 16 + l15) * HW_;
  v8f acc = {};
  for (int k0 = 0; k0 < HW_; k0 += 32)
    acc = wmma_bf16(load_a_frag(arow, k0, h), load_b_frag(brow, k0, h), acc);
  float* out = attn + (size_t)bh * HD_ * HD_;
#pragma unroll
  for (int j = 0; j < 8; ++j)
    out[(tm * 16 + j + 8 * h) * HD_ + tn * 16 + l15] = acc[j];
}

// ---------------- softmax over rows of 48, emit bf16 padded [48][64] ----------------
__global__ void softmax_rows(const float* __restrict__ attn, const float* __restrict__ temp,
                             bf16* __restrict__ attn_bf) {
  __shared__ float s[64];
  int r = blockIdx.x;              // 0 .. B*HEADS*HD-1
  int t = threadIdx.x;             // 0..63
  int bh = r / HD_;
  int hidx = bh % HEADS_;
  const float* row = attn + (size_t)r * HD_;
  float v = (t < HD_) ? row[t] * temp[hidx] : -3.402823e38f;
  s[t] = v; __syncthreads();
  for (int off = 32; off > 0; off >>= 1) {
    if (t < off) s[t] = fmaxf(s[t], s[t + off]);
    __syncthreads();
  }
  float m = s[0]; __syncthreads();
  float e = (t < HD_) ? __expf(v - m) : 0.0f;
  s[t] = e; __syncthreads();
  for (int off = 32; off > 0; off >>= 1) {
    if (t < off) s[t] += s[t + off];
    __syncthreads();
  }
  float inv = 1.0f / s[0];
  attn_bf[(size_t)r * 64 + t] = (t < HD_) ? (bf16)(e * inv) : (bf16)0.0f;
}

// ---------------- out = attn * v per (b,head): M=48, N=9216, K=64(padded) ----------------
// Each wave: full M=48 x one 16-wide N tile; B fragments reused for all three M tiles.
__global__ void av_gemm(const bf16* __restrict__ attn_bf, const bf16* __restrict__ vT,
                        float* __restrict__ tmp) {
  int bh = blockIdx.y;
  int b = bh >> 3, head = bh & 7;
  int wave = threadIdx.x >> 5, lane = threadIdx.x & 31;
  int nt = blockIdx.x * 8 + wave;          // 0..575
  int l15 = lane & 15, h = lane >> 4;

  const bf16* abase = attn_bf + (size_t)bh * HD_ * 64;
  v16bf a[3][2];
#pragma unroll
  for (int mi = 0; mi < 3; ++mi)
#pragma unroll
    for (int kk = 0; kk < 2; ++kk)
      a[mi][kk] = load_a_frag(abase + (size_t)(mi * 16 + l15) * 64, kk * 32, h);

  const bf16* brow = vT + ((size_t)bh * HW_ + nt * 16 + l15) * 64;
  v16bf fb0 = load_b_frag(brow, 0, h);
  v16bf fb1 = load_b_frag(brow, 32, h);

  v8f acc[3] = {};
#pragma unroll
  for (int mi = 0; mi < 3; ++mi) {
    acc[mi] = wmma_bf16(a[mi][0], fb0, acc[mi]);
    acc[mi] = wmma_bf16(a[mi][1], fb1, acc[mi]);
  }
  float* obase = tmp + ((size_t)b * C_ + head * HD_) * HW_;
#pragma unroll
  for (int mi = 0; mi < 3; ++mi)
#pragma unroll
    for (int j = 0; j < 8; ++j)
      obase[(size_t)(mi * 16 + j + 8 * h) * HW_ + nt * 16 + l15] = acc[mi][j];
}

// ---------------- host launcher ----------------

extern "C" void kernel_launch(void* const* d_in, const int* in_sizes, int n_in,
                              void* d_out, int out_size, void* d_ws, size_t ws_size,
                              hipStream_t stream) {
  const float* x      = (const float*)d_in[0];
  const float* w_qkv  = (const float*)d_in[1];
  const float* w_dw   = (const float*)d_in[2];
  const float* w_proj = (const float*)d_in[3];
  const float* temp   = (const float*)d_in[4];
  float* out = (float*)d_out;

  char* ws = (char*)d_ws;
  size_t off = 0;
  auto take = [&](size_t bytes) -> char* {
    char* p = ws + off;
    off = (off + bytes + 255) & ~(size_t)255;
    return p;
  };
  bf16*  xT       = (bf16*)take((size_t)B_ * HW_ * C_ * 2);
  bf16*  wqkv_bf  = (bf16*)take((size_t)C3_ * C_ * 2);
  bf16*  wproj_bf = (bf16*)take((size_t)C_ * C_ * 2);
  float* qkv      = (float*)take((size_t)B_ * C3_ * HW_ * 4);
  float* qkv_dw   = (float*)take((size_t)B_ * C3_ * HW_ * 4);
  bf16*  qbf      = (bf16*)take((size_t)B_ * HEADS_ * HD_ * HW_ * 2);
  bf16*  kbf      = (bf16*)take((size_t)B_ * HEADS_ * HD_ * HW_ * 2);
  bf16*  vT       = (bf16*)take((size_t)B_ * HEADS_ * HW_ * 64 * 2);
  float* invn     = (float*)take((size_t)2 * B_ * HEADS_ * HD_ * 4);
  float* attn     = (float*)take((size_t)B_ * HEADS_ * HD_ * HD_ * 4);
  bf16*  attn_bf  = (bf16*)take((size_t)B_ * HEADS_ * HD_ * 64 * 2);
  // qkv (pre-dwconv) is dead after dwconv3x3 -> reuse its region:
  float* tmp  = qkv;                                             // [B][C][HW] f32
  bf16*  tmpT = (bf16*)((char*)qkv + (size_t)B_ * C_ * HW_ * 4); // [B][HW][C] bf16

  const long nX = (long)B_ * C_ * HW_;

  // 1. x -> x^T bf16 ; weights -> bf16
  cvt_transpose<<<(int)((nX + 255) / 256), 256, 0, stream>>>(x, xT, C_, HW_, nX);
  cvt_cast<<<(C3_ * C_ + 255) / 256, 256, 0, stream>>>(w_qkv, wqkv_bf, (long)C3_ * C_);
  cvt_cast<<<(C_ * C_ + 255) / 256, 256, 0, stream>>>(w_proj, wproj_bf, (long)C_ * C_);

  // 2. qkv = w_qkv @ x   (M=1152, N=9216, K=384, batched over B)
  {
    int tiles = (C3_ / 32) * (HW_ / 32);
    dim3 g((tiles + 7) / 8, B_);
    gemm_bf16<<<g, 256, 0, stream>>>(wqkv_bf, xT, qkv, C3_, HW_, C_,
                                     (long)HW_ * C_, (long)C3_ * HW_);
  }

  // 3. depthwise 3x3
  {
    long n = (long)B_ * C3_ * HW_;
    dwconv3x3<<<(int)((n + 255) / 256), 256, 0, stream>>>(qkv, w_dw, qkv_dw);
  }

  // 4. L2 norms of q,k rows; pack q,k (scaled bf16) and v^T (bf16, K padded to 64)
  rownorm<<<2 * B_ * HEADS_ * HD_, 256, 0, stream>>>(qkv_dw, invn);
  {
    long n = 2L * B_ * HEADS_ * HD_ * HW_;
    pack_qk<<<(int)((n + 255) / 256), 256, 0, stream>>>(qkv_dw, invn, qbf, kbf);
  }
  {
    long n = (long)B_ * HEADS_ * HW_ * 64;
    pack_vT<<<(int)((n + 255) / 256), 256, 0, stream>>>(qkv_dw, vT);
  }

  // 5. attn = q k^T  (per (b,head): 48x48, K=9216)
  attn_gemm<<<B_ * HEADS_, 288, 0, stream>>>(qbf, kbf, attn);

  // 6. softmax(attn * temperature) -> bf16 padded [48][64]
  softmax_rows<<<B_ * HEADS_ * HD_, 64, 0, stream>>>(attn, temp, attn_bf);

  // 7. tmp = attn @ v  (per (b,head): 48 x 9216, K=64)
  av_gemm<<<dim3(HW_ / 16 / 8, B_ * HEADS_), 256, 0, stream>>>(attn_bf, vT, tmp);

  // 8. tmp -> tmp^T bf16 ; out = w_proj @ tmp
  cvt_transpose<<<(int)((nX + 255) / 256), 256, 0, stream>>>(tmp, tmpT, C_, HW_, nX);
  {
    int tiles = (C_ / 32) * (HW_ / 32);
    dim3 g((tiles + 7) / 8, B_);
    gemm_bf16<<<g, 256, 0, stream>>>(wproj_bf, tmpT, out, C_, HW_, C_,
                                     (long)HW_ * C_, (long)C_ * HW_);
  }

  (void)in_sizes; (void)n_in; (void)out_size; (void)ws_size;
}
